// GO_GVP_953482740112
// MI455X (gfx1250) — matrix operand order repaired and checked by reading, not compiled, not run on hardware
//
#include <hip/hip_runtime.h>
#include <hip/hip_bf16.h>
#include <stdint.h>

// ---------------------------------------------------------------------------
// GVP-GNN forward for MI455X (gfx1250).  bf16 WMMA (v_wmma_f32_16x16x32_bf16)
// for all the 200-output GVP scalar GEMMs; VALU for the small vector-channel
// einsums, LayerNorms, gathers/scatters; f32 atomics for segment sums.
// A-fragments are loaded from LDS as 2x ds_load_b128 per WMMA (16B aligned),
// B-fragments as 2x global_load_b128 from a prepacked bf16 fragment buffer.
// ---------------------------------------------------------------------------

typedef __attribute__((ext_vector_type(16))) __bf16        v16bf;
typedef __attribute__((ext_vector_type(8)))  float         v8f;
typedef __attribute__((ext_vector_type(4)))  unsigned int  v4u;

#define NSC   200      // node scalar channels
#define NVC   32       // node vector channels (96 floats/node)
#define ESC   64       // edge scalar channels
#define NTILES 13      // 208 = ceil(200/16) output tiles

// message GVP Ws:  [200, 529]  K padded to 544 (17 tiles of 32)
#define MSG_KP 544
#define MSG_KT 17
// ff / out GVP Ws: [200, 232]  K padded to 256 (8 tiles of 32)
#define FF_KP  256
#define FF_KT  8

__device__ __forceinline__ unsigned short f2bf(float f) {
  union { float f; uint32_t u; } x; x.f = f;
  uint32_t r = x.u + 0x7FFFu + ((x.u >> 16) & 1u);   // round-to-nearest-even
  return (unsigned short)(r >> 16);
}

// K-loop over one 16x16 output tile.
//   arow : LDS bf16 pointer to this lane's A row (row = m0 + (lane&15)), rows
//          stored row-major, 16B-aligned, with each 32-elem K tile contiguous.
//   bbase: prepacked global B fragments for this n-tile:
//          layout [ktile][lane][16 bf16] (32 contiguous bytes per lane).
// A layout (matches B): lanes 0-15 -> K {0..15},
//                       lanes 16-31 -> K {16..31}.
// B layout (row striped across lanes): lane -> n = lane&15,
//                                element e -> K = e + 16*(lane>=16).
__device__ __forceinline__ v8f wmma_kloop(const unsigned short* arow,
                                          const uint32_t* bbase,
                                          int ktiles, int lane, v8f acc) {
  const int q1 = (lane < 16) ? 0 : 2;   // 16-byte chunk indices in the K tile
  const int q2 = (lane < 16) ? 1 : 3;
  for (int kt = 0; kt < ktiles; ++kt) {
    union { v16bf v; v4u q[2]; } a, b;
    const v4u* ap = (const v4u*)(arow + kt * 32);       // -> ds_load_b128 x2
    a.q[0] = ap[q1];
    a.q[1] = ap[q2];
    const v4u* bp = (const v4u*)(bbase + (size_t)(kt * 32 + lane) * 8);
    b.q[0] = bp[0];                                     // -> global_load_b128 x2
    b.q[1] = bp[1];
    acc = __builtin_amdgcn_wmma_f32_16x16x32_bf16(false, a.v, false, b.v,
                                                  (short)0, acc, false, false);
  }
  return acc;
}

// ---------------------------------------------------------------------------
// utility kernels
// ---------------------------------------------------------------------------
__global__ void fill_zero(float* p, int n) {
  int i = blockIdx.x * blockDim.x + threadIdx.x;
  if (i < n) p[i] = 0.f;
}

__global__ void deg_kernel(const int* dst, float* deg, int nE) {
  int i = blockIdx.x * blockDim.x + threadIdx.x;
  if (i < nE) atomicAdd(&deg[dst[i]], 1.f);
}

// Prepack W[N,K] (f32, row-major) into bf16 B fragments:
// out[((ntile*ktiles + ktile)*32 + lane)*16 + e] = W[ntile*16 + (lane&15)]
//                                                   [ktile*32 + e + 16*(lane>=16)]
__global__ void prepack_b(const float* W, int N, int K, int ktiles, int ntiles,
                          unsigned short* out) {
  int idx = blockIdx.x * blockDim.x + threadIdx.x;
  int total = ntiles * ktiles * 512;
  if (idx >= total) return;
  int e    = idx & 15;
  int lane = (idx >> 4) & 31;
  int rest = idx >> 9;
  int kt   = rest % ktiles;
  int nt   = rest / ktiles;
  int n = nt * 16 + (lane & 15);
  int k = kt * 32 + e + ((lane < 16) ? 0 : 16);
  float v = (n < N && k < K) ? W[(size_t)n * K + k] : 0.f;
  out[idx] = f2bf(v);
}

// ---------------------------------------------------------------------------
// embedding GVPs (small dims -> VALU)
// ---------------------------------------------------------------------------
__global__ void emb_node_kernel(const float* node_s, const float* node_V,
                                const float* Wh, const float* Ws,
                                const float* Wv, const float* bs,
                                float* s_out, float* V_out) {
  int n = blockIdx.x, t = threadIdx.x;
  __shared__ float Vh[96];
  __shared__ float vn[32];
  __shared__ float sv[6];
  const float* V = node_V + (size_t)n * 9;
  if (t < 96) {
    int i = t / 3, c = t % 3;
    float a = 0.f;
    for (int v = 0; v < 3; ++v) a += Wh[i * 3 + v] * V[v * 3 + c];
    Vh[t] = a;
  }
  if (t < 6) sv[t] = node_s[(size_t)n * 6 + t];
  __syncthreads();
  if (t < 32)
    vn[t] = sqrtf(Vh[t*3]*Vh[t*3] + Vh[t*3+1]*Vh[t*3+1] + Vh[t*3+2]*Vh[t*3+2] + 1e-8f);
  __syncthreads();
  if (t < NSC) {
    float a = bs[t];
    for (int k = 0; k < 6;  ++k) a += Ws[t * 38 + k]     * sv[k];
    for (int k = 0; k < 32; ++k) a += Ws[t * 38 + 6 + k] * vn[k];
    s_out[(size_t)n * NSC + t] = fmaxf(a, 0.f);
  }
  if (t < 32) {
    float a0 = 0.f, a1 = 0.f, a2 = 0.f;
    for (int h = 0; h < 32; ++h) {
      float w = Wv[t * 32 + h];
      a0 += w * Vh[h*3]; a1 += w * Vh[h*3+1]; a2 += w * Vh[h*3+2];
    }
    float un = sqrtf(a0*a0 + a1*a1 + a2*a2 + 1e-8f);
    float g  = 1.f / (1.f + expf(-un));
    float* o = V_out + (size_t)n * 96 + t * 3;
    o[0] = a0 * g; o[1] = a1 * g; o[2] = a2 * g;
  }
}

__global__ void emb_edge_kernel(const float* edge_s, const float* edge_V,
                                const float* Wh, const float* Ws,
                                const float* Wv, const float* bs,
                                float* es_out, float* eV_out, int nE) {
  int idx = blockIdx.x * blockDim.x + threadIdx.x;
  if (idx >= nE * ESC) return;
  int e = idx / ESC, j = idx % ESC;
  const float* Vv = edge_V + (size_t)e * 3;
  float wh = Wh[0];
  float v0 = wh * Vv[0], v1 = wh * Vv[1], v2 = wh * Vv[2];
  float vn = sqrtf(v0*v0 + v1*v1 + v2*v2 + 1e-8f);
  float a = bs[j];
  const float* sr = edge_s + (size_t)e * 32;
  for (int k = 0; k < 32; ++k) a += Ws[j * 33 + k] * sr[k];
  a += Ws[j * 33 + 32] * vn;
  es_out[(size_t)e * ESC + j] = fmaxf(a, 0.f);
  if (j == 0) {
    float wv = Wv[0];
    float u0 = wv * v0, u1 = wv * v1, u2 = wv * v2;
    float un = sqrtf(u0*u0 + u1*u1 + u2*u2 + 1e-8f);
    float g  = 1.f / (1.f + expf(-un));
    float* o = eV_out + (size_t)e * 3;
    o[0] = u0 * g; o[1] = u1 * g; o[2] = u2 * g;
  }
}

// ---------------------------------------------------------------------------
// message GVP: 64 edges/block, WMMA for the [E,529]x[529,200] GEMM
// ---------------------------------------------------------------------------
__global__ __launch_bounds__(256)
void msg_kernel(const float* s, const float* V, const float* es, const float* eV,
                const int* src, const int* dst,
                const float* Wh, const unsigned short* WsB,
                const float* Wv, const float* bs,
                float* dsAcc, float* dVAcc, int nE) {
  extern __shared__ char smem[];
  unsigned short* feat = (unsigned short*)smem;                 // [64][544] bf16
  float* mV = (float*)(smem + 64 * MSG_KP * 2);                 // [64][65][3]
  float* Vh = mV + 64 * 195;                                    // [64][65][3]
  int tid = threadIdx.x;
  int e0  = blockIdx.x * 64;

  // scalar features 0..463 : [s[src] | es | s[dst]]
  for (int idx = tid; idx < 64 * 464; idx += 256) {
    int e = idx / 464, f = idx % 464, ge = e0 + e;
    float v = 0.f;
    if (ge < nE) {
      if (f < 200)      v = s[(size_t)src[ge] * NSC + f];
      else if (f < 264) v = es[(size_t)ge * ESC + (f - 200)];
      else              v = s[(size_t)dst[ge] * NSC + (f - 264)];
    }
    feat[e * MSG_KP + f] = f2bf(v);
  }
  // gather vector inputs [V[src] | eV | V[dst]] -> 65 vectors of 3
  for (int idx = tid; idx < 64 * 195; idx += 256) {
    int e = idx / 195, r = idx % 195, ge = e0 + e;
    float v = 0.f;
    if (ge < nE) {
      int vi = r / 3, c = r % 3;
      if (vi < 32)       v = V[(size_t)src[ge] * 96 + vi * 3 + c];
      else if (vi == 32) v = eV[(size_t)ge * 3 + c];
      else               v = V[(size_t)dst[ge] * 96 + (vi - 33) * 3 + c];
    }
    mV[idx] = v;
  }
  __syncthreads();
  // Vh = Wh[65,65] @ mV
  for (int idx = tid; idx < 64 * 195; idx += 256) {
    int e = idx / 195, r = idx % 195, i = r / 3, c = r % 3;
    const float* me = mV + e * 195;
    float a = 0.f;
    for (int v = 0; v < 65; ++v) a += Wh[i * 65 + v] * me[v * 3 + c];
    Vh[idx] = a;
  }
  __syncthreads();
  // features 464..528 = |Vh|, pad 529..543 with zero
  for (int idx = tid; idx < 64 * 80; idx += 256) {
    int e = idx / 80, i = idx % 80;
    float v = 0.f;
    if (i < 65) {
      const float* p = Vh + e * 195 + i * 3;
      v = sqrtf(p[0]*p[0] + p[1]*p[1] + p[2]*p[2] + 1e-8f);
    }
    feat[e * MSG_KP + 464 + i] = f2bf(v);
  }
  __syncthreads();

  // WMMA GEMM: 4 m-tiles x 13 n-tiles, fused bias+ReLU+scatter
  int wave = tid >> 5, lane = tid & 31;
  for (int tile = wave; tile < 4 * NTILES; tile += 8) {
    int mt = tile / NTILES, nt = tile % NTILES;
    v8f acc = {0.f,0.f,0.f,0.f,0.f,0.f,0.f,0.f};
    const unsigned short* arow = feat + (size_t)(mt * 16 + (lane & 15)) * MSG_KP;
    const uint32_t* bbase = (const uint32_t*)WsB + (size_t)nt * MSG_KT * 256;
    acc = wmma_kloop(arow, bbase, MSG_KT, lane, acc);
    int n = nt * 16 + (lane & 15);
    if (n < NSC) {
      float bias = bs[n];
#pragma unroll
      for (int r = 0; r < 8; ++r) {
        int m  = r + ((lane < 16) ? 0 : 8);
        int ge = e0 + mt * 16 + m;
        if (ge < nE) {
          float val = fmaxf(acc[r] + bias, 0.f);
          atomicAdd(&dsAcc[(size_t)dst[ge] * NSC + n], val);
        }
      }
    }
  }
  // vector messages: Vu = Wv[32,65] @ Vh, sigmoid-gated, scattered
  for (int idx = tid; idx < 64 * 32; idx += 256) {
    int e = idx / 32, u = idx % 32, ge = e0 + e;
    if (ge >= nE) continue;
    const float* p = Vh + e * 195;
    float a0 = 0.f, a1 = 0.f, a2 = 0.f;
    for (int h = 0; h < 65; ++h) {
      float w = Wv[u * 65 + h];
      a0 += w * p[h*3]; a1 += w * p[h*3+1]; a2 += w * p[h*3+2];
    }
    float un = sqrtf(a0*a0 + a1*a1 + a2*a2 + 1e-8f);
    float g  = 1.f / (1.f + expf(-un));
    float* o = dVAcc + (size_t)dst[ge] * 96 + u * 3;
    atomicAdd(o + 0, a0 * g);
    atomicAdd(o + 1, a1 * g);
    atomicAdd(o + 2, a2 * g);
  }
}

// ---------------------------------------------------------------------------
// node update: residual + LN1 + ff-GVP (WMMA) + LN2.  64 nodes/block.
// ---------------------------------------------------------------------------
__global__ __launch_bounds__(256)
void node_update_kernel(float* s, float* V, const float* dsAcc, const float* dVAcc,
                        const float* deg,
                        const float* ln1g, const float* ln1b,
                        const float* ln2g, const float* ln2b,
                        const float* Wh, const unsigned short* WsB,
                        const float* Wv, const float* bs, int nN) {
  extern __shared__ char smem[];
  float* sbuf = (float*)smem;                 // [64][200]
  float* vbuf = sbuf + 64 * NSC;              // [64][96]
  float* vh2  = vbuf + 64 * 96;               // [64][96]
  float* stat = vh2  + 64 * 96;               // [64][4]  mu, rstd, vfac
  unsigned short* feat = (unsigned short*)(stat + 64 * 4);  // [64][256] bf16
  int tid = threadIdx.x, n0 = blockIdx.x * 64;

  for (int idx = tid; idx < 64 * NSC; idx += 256) {
    int e = idx / NSC, j = idx % NSC, gn = n0 + e;
    float v = 0.f;
    if (gn < nN) {
      float d = fmaxf(deg[gn], 1.f);
      v = s[(size_t)gn * NSC + j] + dsAcc[(size_t)gn * NSC + j] / d;
    }
    sbuf[idx] = v;
  }
  for (int idx = tid; idx < 64 * 96; idx += 256) {
    int e = idx / 96, gn = n0 + e;
    float v = 0.f;
    if (gn < nN) {
      float d = fmaxf(deg[gn], 1.f);
      v = V[(size_t)gn * 96 + idx % 96] + dVAcc[(size_t)gn * 96 + idx % 96] / d;
    }
    vbuf[idx] = v;
  }
  __syncthreads();
  if (tid < 64) {
    float mu = 0.f;
    for (int j = 0; j < NSC; ++j) mu += sbuf[tid * NSC + j];
    mu /= (float)NSC;
    float var = 0.f;
    for (int j = 0; j < NSC; ++j) { float d = sbuf[tid * NSC + j] - mu; var += d * d; }
    var /= (float)NSC;
    stat[tid * 4 + 0] = mu;
    stat[tid * 4 + 1] = rsqrtf(var + 1e-5f);
  } else if (tid < 128) {
    int e = tid - 64;
    float vs = 0.f;
    for (int j = 0; j < 96; ++j) { float x = vbuf[e * 96 + j]; vs += x * x; }
    stat[e * 4 + 2] = rsqrtf(vs / 32.f + 1e-8f);
  }
  __syncthreads();
  for (int idx = tid; idx < 64 * NSC; idx += 256) {
    int e = idx / NSC, j = idx % NSC;
    float x = (sbuf[idx] - stat[e*4]) * stat[e*4+1] * ln1g[j] + ln1b[j];
    sbuf[idx] = x;
    feat[e * FF_KP + j] = f2bf(x);
  }
  for (int idx = tid; idx < 64 * 96; idx += 256) {
    int e = idx / 96;
    vbuf[idx] *= stat[e * 4 + 2];
  }
  __syncthreads();
  // Vh2 = Wh_ff[32,32] @ V
  for (int idx = tid; idx < 64 * 96; idx += 256) {
    int e = idx / 96, r = idx % 96, u = r / 3, c = r % 3;
    const float* vb = vbuf + e * 96;
    float a = 0.f;
    for (int h = 0; h < 32; ++h) a += Wh[u * 32 + h] * vb[h * 3 + c];
    vh2[idx] = a;
  }
  __syncthreads();
  for (int idx = tid; idx < 64 * 56; idx += 256) {
    int e = idx / 56, i = idx % 56;
    float v = 0.f;
    if (i < 32) {
      const float* p = vh2 + e * 96 + i * 3;
      v = sqrtf(p[0]*p[0] + p[1]*p[1] + p[2]*p[2] + 1e-8f);
    }
    feat[e * FF_KP + 200 + i] = f2bf(v);
  }
  __syncthreads();
  // WMMA ff GEMM + residual into sbuf
  int wave = tid >> 5, lane = tid & 31;
  for (int tile = wave; tile < 4 * NTILES; tile += 8) {
    int mt = tile / NTILES, nt = tile % NTILES;
    v8f acc = {0.f,0.f,0.f,0.f,0.f,0.f,0.f,0.f};
    const unsigned short* arow = feat + (size_t)(mt * 16 + (lane & 15)) * FF_KP;
    const uint32_t* bbase = (const uint32_t*)WsB + (size_t)nt * FF_KT * 256;
    acc = wmma_kloop(arow, bbase, FF_KT, lane, acc);
    int n = nt * 16 + (lane & 15);
    if (n < NSC) {
      float bias = bs[n];
#pragma unroll
      for (int r = 0; r < 8; ++r) {
        int m = r + ((lane < 16) ? 0 : 8);
        sbuf[(mt * 16 + m) * NSC + n] += fmaxf(acc[r] + bias, 0.f);
      }
    }
  }
  // fV = Wv_ff @ Vh2, gated, residual into vbuf
  for (int idx = tid; idx < 64 * 32; idx += 256) {
    int e = idx / 32, u = idx % 32;
    const float* p = vh2 + e * 96;
    float a0 = 0.f, a1 = 0.f, a2 = 0.f;
    for (int h = 0; h < 32; ++h) {
      float w = Wv[u * 32 + h];
      a0 += w * p[h*3]; a1 += w * p[h*3+1]; a2 += w * p[h*3+2];
    }
    float un = sqrtf(a0*a0 + a1*a1 + a2*a2 + 1e-8f);
    float g  = 1.f / (1.f + expf(-un));
    vbuf[e*96 + u*3 + 0] += a0 * g;
    vbuf[e*96 + u*3 + 1] += a1 * g;
    vbuf[e*96 + u*3 + 2] += a2 * g;
  }
  __syncthreads();
  // LN2 stats
  if (tid < 64) {
    float mu = 0.f;
    for (int j = 0; j < NSC; ++j) mu += sbuf[tid * NSC + j];
    mu /= (float)NSC;
    float var = 0.f;
    for (int j = 0; j < NSC; ++j) { float d = sbuf[tid * NSC + j] - mu; var += d * d; }
    var /= (float)NSC;
    stat[tid * 4 + 0] = mu;
    stat[tid * 4 + 1] = rsqrtf(var + 1e-5f);
  } else if (tid < 128) {
    int e = tid - 64;
    float vs = 0.f;
    for (int j = 0; j < 96; ++j) { float x = vbuf[e * 96 + j]; vs += x * x; }
    stat[e * 4 + 2] = rsqrtf(vs / 32.f + 1e-8f);
  }
  __syncthreads();
  for (int idx = tid; idx < 64 * NSC; idx += 256) {
    int e = idx / NSC, j = idx % NSC, gn = n0 + e;
    if (gn < nN)
      s[(size_t)gn * NSC + j] = (sbuf[idx] - stat[e*4]) * stat[e*4+1] * ln2g[j] + ln2b[j];
  }
  for (int idx = tid; idx < 64 * 96; idx += 256) {
    int e = idx / 96, gn = n0 + e;
    if (gn < nN) V[(size_t)gn * 96 + idx % 96] = vbuf[idx] * stat[e * 4 + 2];
  }
}

// ---------------------------------------------------------------------------
// output GVP (no act) + graph readout scatter.  64 nodes/block, WMMA.
// ---------------------------------------------------------------------------
__global__ __launch_bounds__(256)
void out_readout_kernel(const float* s, const float* V, const int* batch,
                        const float* Wh, const unsigned short* WsB,
                        const float* bs, float* readout, int nN) {
  extern __shared__ char smem[];
  float* vbuf = (float*)smem;                   // [64][96]
  float* vh   = vbuf + 64 * 96;                 // [64][96]
  unsigned short* feat = (unsigned short*)(vh + 64 * 96);  // [64][256]
  int tid = threadIdx.x, n0 = blockIdx.x * 64;

  for (int idx = tid; idx < 64 * NSC; idx += 256) {
    int e = idx / NSC, j = idx % NSC, gn = n0 + e;
    feat[e * FF_KP + j] = f2bf(gn < nN ? s[(size_t)gn * NSC + j] : 0.f);
  }
  for (int idx = tid; idx < 64 * 96; idx += 256) {
    int e = idx / 96, gn = n0 + e;
    vbuf[idx] = gn < nN ? V[(size_t)gn * 96 + idx % 96] : 0.f;
  }
  __syncthreads();
  for (int idx = tid; idx < 64 * 96; idx += 256) {
    int e = idx / 96, r = idx % 96, u = r / 3, c = r % 3;
    const float* vb = vbuf + e * 96;
    float a = 0.f;
    for (int h = 0; h < 32; ++h) a += Wh[u * 32 + h] * vb[h * 3 + c];
    vh[idx] = a;
  }
  __syncthreads();
  for (int idx = tid; idx < 64 * 56; idx += 256) {
    int e = idx / 56, i = idx % 56;
    float v = 0.f;
    if (i < 32) {
      const float* p = vh + e * 96 + i * 3;
      v = sqrtf(p[0]*p[0] + p[1]*p[1] + p[2]*p[2] + 1e-8f);
    }
    feat[e * FF_KP + 200 + i] = f2bf(v);
  }
  __syncthreads();
  int wave = tid >> 5, lane = tid & 31;
  for (int tile = wave; tile < 4 * NTILES; tile += 8) {
    int mt = tile / NTILES, nt = tile % NTILES;
    v8f acc = {0.f,0.f,0.f,0.f,0.f,0.f,0.f,0.f};
    const unsigned short* arow = feat + (size_t)(mt * 16 + (lane & 15)) * FF_KP;
    const uint32_t* bbase = (const uint32_t*)WsB + (size_t)nt * FF_KT * 256;
    acc = wmma_kloop(arow, bbase, FF_KT, lane, acc);
    int n = nt * 16 + (lane & 15);
    if (n < NSC) {
      float bias = bs[n];
#pragma unroll
      for (int r = 0; r < 8; ++r) {
        int m  = r + ((lane < 16) ? 0 : 8);
        int gn = n0 + mt * 16 + m;
        if (gn < nN)
          atomicAdd(&readout[(size_t)batch[gn] * NSC + n], acc[r] + bias);
      }
    }
  }
}

// ---------------------------------------------------------------------------
// tiny dense MLP (64 rows)
// ---------------------------------------------------------------------------
__global__ void dense_kernel(const float* in, const float* W, const float* b,
                             float* out, int rows, int din, int dout, int relu) {
  int idx = blockIdx.x * blockDim.x + threadIdx.x;
  if (idx >= rows * dout) return;
  int r = idx / dout, o = idx % dout;
  const float* x = in + (size_t)r * din;
  const float* w = W  + (size_t)o * din;
  float a = b[o];
  for (int k = 0; k < din; ++k) a += w[k] * x[k];
  out[idx] = relu ? fmaxf(a, 0.f) : a;
}

// ---------------------------------------------------------------------------
// host launch
// ---------------------------------------------------------------------------
extern "C" void kernel_launch(void* const* d_in, const int* in_sizes, int n_in,
                              void* d_out, int out_size, void* d_ws, size_t ws_size,
                              hipStream_t stream) {
  (void)ws_size; (void)n_in;
  const int nN = in_sizes[0] / 6;     // 20000 nodes
  const int nE = in_sizes[2] / 32;    // 320000 edges
  const int nG = out_size / 489;      // 64 graphs

  // input leaves: top-level dict insertion order, params flattened with
  // sorted dict keys (jax pytree order).
  const float* node_s = (const float*)d_in[0];
  const float* node_V = (const float*)d_in[1];
  const float* edge_s = (const float*)d_in[2];
  const float* edge_V = (const float*)d_in[3];
  const float* d1W = (const float*)d_in[4];  const float* d1b = (const float*)d_in[5];
  const float* d2W = (const float*)d_in[6];  const float* d2b = (const float*)d_in[7];
  const float* d3W = (const float*)d_in[8];  const float* d3b = (const float*)d_in[9];
  const float* eWh = (const float*)d_in[10]; const float* eWs = (const float*)d_in[11];
  const float* eWv = (const float*)d_in[12]; const float* ebs = (const float*)d_in[13];
  const float* nWh = (const float*)d_in[14]; const float* nWs = (const float*)d_in[15];
  const float* nWv = (const float*)d_in[16]; const float* nbs = (const float*)d_in[17];
  // layers: base 18 + 12*l: ff.Wh, ff.Ws, ff.Wv, ff.bs, ln1_b, ln1_g, ln2_b, ln2_g,
  //                         msg.Wh, msg.Ws, msg.Wv, msg.bs
  const float* oWh = (const float*)d_in[54];
  const float* oWs = (const float*)d_in[55];
  const float* obs = (const float*)d_in[56];
  const int* edge_index = (const int*)d_in[57];
  const int* batch      = (const int*)d_in[58];
  const int* srcI = edge_index;
  const int* dstI = edge_index + nE;

  // workspace carve
  char* p = (char*)d_ws;
  auto alloc = [&](size_t bytes) -> void* {
    void* r = (void*)p;
    p += (bytes + 255) & ~(size_t)255;
    return r;
  };
  float* deg   = (float*)alloc((size_t)nN * 4);
  float* sbuf  = (float*)alloc((size_t)nN * NSC * 4);
  float* Vbuf  = (float*)alloc((size_t)nN * 96 * 4);
  float* esb   = (float*)alloc((size_t)nE * ESC * 4);
  float* eVb   = (float*)alloc((size_t)nE * 3 * 4);
  float* dsAcc = (float*)alloc((size_t)nN * NSC * 4);
  float* dVAcc = (float*)alloc((size_t)nN * 96 * 4);
  float* rdout = (float*)alloc((size_t)nG * NSC * 4);
  float* h1    = (float*)alloc((size_t)nG * 400 * 4);
  float* h2    = (float*)alloc((size_t)nG * 800 * 4);
  const size_t msgB = (size_t)NTILES * MSG_KT * 512 * 2;
  const size_t ffB  = (size_t)NTILES * FF_KT  * 512 * 2;
  unsigned short* Bmsg[3]; unsigned short* Bff[3];
  for (int l = 0; l < 3; ++l) Bmsg[l] = (unsigned short*)alloc(msgB);
  for (int l = 0; l < 3; ++l) Bff[l]  = (unsigned short*)alloc(ffB);
  unsigned short* Bout = (unsigned short*)alloc(ffB);

  auto zero = [&](float* q, int n) {
    fill_zero<<<(n + 255) / 256, 256, 0, stream>>>(q, n);
  };

  // weight prepack (bf16 B-fragment layout)
  for (int l = 0; l < 3; ++l) {
    const float* msgWs = (const float*)d_in[18 + 12 * l + 9];
    const float* ffWs  = (const float*)d_in[18 + 12 * l + 1];
    int tm = NTILES * MSG_KT * 512, tf = NTILES * FF_KT * 512;
    prepack_b<<<(tm + 255) / 256, 256, 0, stream>>>(msgWs, NSC, 529, MSG_KT, NTILES, Bmsg[l]);
    prepack_b<<<(tf + 255) / 256, 256, 0, stream>>>(ffWs,  NSC, 232, FF_KT,  NTILES, Bff[l]);
  }
  {
    int tf = NTILES * FF_KT * 512;
    prepack_b<<<(tf + 255) / 256, 256, 0, stream>>>(oWs, NSC, 232, FF_KT, NTILES, Bout);
  }

  // degrees
  zero(deg, nN);
  deg_kernel<<<(nE + 255) / 256, 256, 0, stream>>>(dstI, deg, nE);

  // embeddings
  emb_node_kernel<<<nN, 256, 0, stream>>>(node_s, node_V, nWh, nWs, nWv, nbs, sbuf, Vbuf);
  emb_edge_kernel<<<(nE * ESC + 255) / 256, 256, 0, stream>>>(
      edge_s, edge_V, eWh, eWs, eWv, ebs, esb, eVb, nE);

  const size_t MSG_LDS  = (size_t)64 * MSG_KP * 2 + (size_t)64 * 195 * 4 * 2;  // 169472
  const size_t NODE_LDS = (size_t)64 * NSC * 4 + (size_t)64 * 96 * 4 * 2
                        + (size_t)64 * 4 * 4 + (size_t)64 * FF_KP * 2;          // 134144
  const size_t OUT_LDS  = (size_t)64 * 96 * 4 * 2 + (size_t)64 * FF_KP * 2;     // 81920
  const int msgBlocks  = (nE + 63) / 64;
  const int nodeBlocks = (nN + 63) / 64;

  for (int l = 0; l < 3; ++l) {
    const float* mWh = (const float*)d_in[18 + 12 * l + 8];
    const float* mWv = (const float*)d_in[18 + 12 * l + 10];
    const float* mbs = (const float*)d_in[18 + 12 * l + 11];
    const float* fWh = (const float*)d_in[18 + 12 * l + 0];
    const float* fWv = (const float*)d_in[18 + 12 * l + 2];
    const float* fbs = (const float*)d_in[18 + 12 * l + 3];
    const float* l1b = (const float*)d_in[18 + 12 * l + 4];
    const float* l1g = (const float*)d_in[18 + 12 * l + 5];
    const float* l2b = (const float*)d_in[18 + 12 * l + 6];
    const float* l2g = (const float*)d_in[18 + 12 * l + 7];

    zero(dsAcc, nN * NSC);
    zero(dVAcc, nN * 96);
    msg_kernel<<<msgBlocks, 256, MSG_LDS, stream>>>(
        sbuf, Vbuf, esb, eVb, srcI, dstI, mWh, Bmsg[l], mWv, mbs, dsAcc, dVAcc, nE);
    node_update_kernel<<<nodeBlocks, 256, NODE_LDS, stream>>>(
        sbuf, Vbuf, dsAcc, dVAcc, deg, l1g, l1b, l2g, l2b, fWh, Bff[l], fWv, fbs, nN);
  }

  // output GVP + readout
  zero(rdout, nG * NSC);
  out_readout_kernel<<<nodeBlocks, 256, OUT_LDS, stream>>>(
      sbuf, Vbuf, batch, oWh, Bout, obs, rdout, nN);

  // dense head
  dense_kernel<<<(nG * 400 + 255) / 256, 256, 0, stream>>>(rdout, d1W, d1b, h1, nG, 200, 400, 1);
  dense_kernel<<<(nG * 800 + 255) / 256, 256, 0, stream>>>(h1,    d2W, d2b, h2, nG, 400, 800, 1);
  dense_kernel<<<(nG * 489 + 255) / 256, 256, 0, stream>>>(h2,    d3W, d3b, (float*)d_out, nG, 800, 489, 0);
}
